// MultiScaleAttention_5308579577934
// MI455X (gfx1250) — matrix operand order repaired and checked by reading, not compiled, not run on hardware
//
#include <hip/hip_runtime.h>
#include <hip/hip_bf16.h>
#include <stdint.h>

#define NHD   12
#define HDD   64
#define DIMM  768
#define NTOK_IN 3137
#define NTOK  785
#define NTOKP 832     // padded token count (multiple of 64) for K-padding
#define GP    28
#define NPIX  784

typedef __attribute__((ext_vector_type(8)))  float  v8f;
typedef __attribute__((ext_vector_type(8)))  __bf16 v8bf;
typedef __attribute__((ext_vector_type(16))) __bf16 v16bf;

#if defined(__has_builtin)
#if __has_builtin(__builtin_amdgcn_cvt_pk_bf16_f32)
#define HAVE_CVT_PK_BF16 1
#endif
#endif

__device__ __forceinline__ __bf16 f2bf(float f) {
  union { float f; uint32_t u; } c; c.f = f;
  uint32_t r = c.u + 0x7FFFu + ((c.u >> 16) & 1u);   // round-to-nearest-even
  union { uint16_t s; __bf16 b; } o; o.s = (uint16_t)(r >> 16);
  return o.b;
}

// pack 8 f32 -> 8 bf16 (hardware pack-convert when available)
__device__ __forceinline__ v8bf pack8(float4 f0, float4 f1) {
  v8bf r;
#ifdef HAVE_CVT_PK_BF16
  typedef __attribute__((ext_vector_type(2))) __bf16 v2bf;
  v2bf p0 = __builtin_amdgcn_cvt_pk_bf16_f32(f0.x, f0.y);
  v2bf p1 = __builtin_amdgcn_cvt_pk_bf16_f32(f0.z, f0.w);
  v2bf p2 = __builtin_amdgcn_cvt_pk_bf16_f32(f1.x, f1.y);
  v2bf p3 = __builtin_amdgcn_cvt_pk_bf16_f32(f1.z, f1.w);
  r[0]=p0[0]; r[1]=p0[1]; r[2]=p1[0]; r[3]=p1[1];
  r[4]=p2[0]; r[5]=p2[1]; r[6]=p3[0]; r[7]=p3[1];
#else
  r[0]=f2bf(f0.x); r[1]=f2bf(f0.y); r[2]=f2bf(f0.z); r[3]=f2bf(f0.w);
  r[4]=f2bf(f1.x); r[5]=f2bf(f1.y); r[6]=f2bf(f1.z); r[7]=f2bf(f1.w);
#endif
  return r;
}

__device__ __forceinline__ v16bf cat8(v8bf a, v8bf b) {
  return __builtin_shufflevector(a, b, 0,1,2,3,4,5,6,7,8,9,10,11,12,13,14,15);
}

// ---------------------------------------------------------------------------
// Batched GEMM: C = alpha * A(f32) x B(f32) [+ bias], bf16 WMMA path.
// A is [M,K] row-major. B_IS_KN=true: B is [K,N] row-major; false: B is [N,K].
// REQUIRES: K % 64 == 0 (callers pad). M/N edges handled by clamped staging
// loads (garbage rows/cols computed but never stored) + guarded epilogue.
// Block tile 128x64x64, 8 waves (4x2), each wave: 32x32, 8 WMMAs per barrier.
// ---------------------------------------------------------------------------
#define BMT 128
#define BNT 64
#define BKT 64

template<bool B_IS_KN>
__global__ __launch_bounds__(256)
void wmma_gemm_kernel(const float* __restrict__ A, const float* __restrict__ Bm,
                      float* __restrict__ C, const float* __restrict__ bias,
                      int M, int N, int K, int lda, int ldb, int ldc,
                      long long sA, long long sB, long long sC, float alpha)
{
  __shared__ __bf16 As[BMT][BKT + 8];   // +8 bf16 pad: 16B-aligned rows, fewer bank conflicts
  __shared__ __bf16 Bs[BNT][BKT + 8];   // stored as [n][k]

  A  += (size_t)blockIdx.z * sA;
  Bm += (size_t)blockIdx.z * sB;
  C  += (size_t)blockIdx.z * sC;

  const int t     = threadIdx.x;
  const int lane  = t & 31;
  const int wave  = t >> 5;
  const int waveM = wave >> 1;    // 0..3  -> 32-row slab
  const int waveN = wave & 1;     // 0..1  -> 32-col slab
  const int lrow  = lane & 15;
  const int hi16  = lane >> 4;    // 0 or 1

  const int m0 = blockIdx.y * BMT;
  const int n0 = blockIdx.x * BNT;

  // A staging map: row = t>>1 (0..127), kk = (t&1)*32 -> 32 consecutive floats
  const int a_row = t >> 1;
  const int a_kk  = (t & 1) * 32;
  int a_gm = m0 + a_row; if (a_gm >= M) a_gm = M - 1;      // clamp (never stored)

  // B staging maps (tile staged as [n][k], 64x64)
  const int bkn_nn = t & 63;            // [K,N]: column
  const int bkn_kk = (t >> 6) * 16;     // [K,N]: k-chunk of 16
  const int bnk_nn = t >> 2;            // [N,K]: row (=column of C)
  const int bnk_kk = (t & 3) * 16;      // [N,K]: k-chunk of 16
  int b_gn = B_IS_KN ? (n0 + bkn_nn) : (n0 + bnk_nn);
  if (b_gn >= N) b_gn = N - 1;                              // clamp (never stored)

  v8f acc00 = {0.f,0.f,0.f,0.f,0.f,0.f,0.f,0.f};
  v8f acc01 = acc00, acc10 = acc00, acc11 = acc00;

  for (int k0 = 0; k0 < K; k0 += BKT) {
    // ---- stage A tile (128 x 64): 8 x global_load_b128 + 4 x ds_store_b128 ----
    {
      const float4* ap4 = (const float4*)(A + (size_t)a_gm * lda + k0 + a_kk);
      float4 f0 = ap4[0], f1 = ap4[1], f2 = ap4[2], f3 = ap4[3];
      float4 f4 = ap4[4], f5 = ap4[5], f6 = ap4[6], f7 = ap4[7];
      *(v8bf*)&As[a_row][a_kk]      = pack8(f0, f1);
      *(v8bf*)&As[a_row][a_kk + 8]  = pack8(f2, f3);
      *(v8bf*)&As[a_row][a_kk + 16] = pack8(f4, f5);
      *(v8bf*)&As[a_row][a_kk + 24] = pack8(f6, f7);
      if (k0 + BKT < K)
        __builtin_prefetch(A + (size_t)a_gm * lda + k0 + BKT + a_kk, 0, 0); // global_prefetch_b8
    }
    // ---- stage B tile as [n][k] (64 x 64) ----
    if (B_IS_KN) {
      // 16 lane-coalesced b32 loads (consecutive lanes -> consecutive n)
      const float* bp = Bm + (size_t)(k0 + bkn_kk) * ldb + b_gn;
      float fb[16];
      #pragma unroll
      for (int i = 0; i < 16; ++i) fb[i] = bp[(size_t)i * ldb];
      float4 g0 = {fb[0],fb[1],fb[2],fb[3]},   g1 = {fb[4],fb[5],fb[6],fb[7]};
      float4 g2 = {fb[8],fb[9],fb[10],fb[11]}, g3 = {fb[12],fb[13],fb[14],fb[15]};
      *(v8bf*)&Bs[bkn_nn][bkn_kk]     = pack8(g0, g1);
      *(v8bf*)&Bs[bkn_nn][bkn_kk + 8] = pack8(g2, g3);
      if (k0 + BKT < K)
        __builtin_prefetch(Bm + (size_t)(k0 + BKT + bkn_kk) * ldb + b_gn, 0, 0);
    } else {
      const float4* bp4 = (const float4*)(Bm + (size_t)b_gn * ldb + k0 + bnk_kk);
      float4 f0 = bp4[0], f1 = bp4[1], f2 = bp4[2], f3 = bp4[3];
      *(v8bf*)&Bs[bnk_nn][bnk_kk]     = pack8(f0, f1);
      *(v8bf*)&Bs[bnk_nn][bnk_kk + 8] = pack8(f2, f3);
      if (k0 + BKT < K)
        __builtin_prefetch(Bm + (size_t)b_gn * ldb + k0 + BKT + bnk_kk, 0, 0);
    }
    __syncthreads();

    // ---- 2 K-substeps of 32; 8 WMMAs per barrier pair ----
    const int ar0 = waveM * 32 + lrow;
    const int ar1 = ar0 + 16;
    const int br0 = waveN * 32 + lrow;
    const int br1 = br0 + 16;
    #pragma unroll
    for (int s = 0; s < 2; ++s) {
      const int ka = s * 32 + hi16 * 16;   // A: lanes0-15 K=0..15, lanes16-31 K=16..31
      const int kb = s * 32 + hi16 * 16;   // B: lanes0-15 K=0..15, lanes16-31 K=16..31

      v16bf a0 = cat8(*(const v8bf*)&As[ar0][ka], *(const v8bf*)&As[ar0][ka + 8]);
      v16bf a1 = cat8(*(const v8bf*)&As[ar1][ka], *(const v8bf*)&As[ar1][ka + 8]);
      v16bf b0 = cat8(*(const v8bf*)&Bs[br0][kb], *(const v8bf*)&Bs[br0][kb + 8]);
      v16bf b1 = cat8(*(const v8bf*)&Bs[br1][kb], *(const v8bf*)&Bs[br1][kb + 8]);

      acc00 = __builtin_amdgcn_wmma_f32_16x16x32_bf16(false, a0, false, b0, (short)0, acc00, false, false);
      acc01 = __builtin_amdgcn_wmma_f32_16x16x32_bf16(false, a0, false, b1, (short)0, acc01, false, false);
      acc10 = __builtin_amdgcn_wmma_f32_16x16x32_bf16(false, a1, false, b0, (short)0, acc10, false, false);
      acc11 = __builtin_amdgcn_wmma_f32_16x16x32_bf16(false, a1, false, b1, (short)0, acc11, false, false);
    }
    __syncthreads();
  }

  // ---- epilogue: C/D layout per ISA (VGPR r -> M=r(+8 for hi lanes), N=lane) ----
  const int gn0 = n0 + waveN * 32 + lrow;
  const int gn1 = gn0 + 16;
  const int gm0 = m0 + waveM * 32 + hi16 * 8;
  const int gm1 = gm0 + 16;

  if (m0 + BMT <= M && n0 + BNT <= N) {          // fast full-tile path (uniform branch)
    float bv0 = bias ? bias[gn0] : 0.f;
    float bv1 = bias ? bias[gn1] : 0.f;
    #pragma unroll
    for (int r = 0; r < 8; ++r) {
      C[(size_t)(gm0 + r) * ldc + gn0] = acc00[r] * alpha + bv0;
      C[(size_t)(gm0 + r) * ldc + gn1] = acc01[r] * alpha + bv1;
      C[(size_t)(gm1 + r) * ldc + gn0] = acc10[r] * alpha + bv0;
      C[(size_t)(gm1 + r) * ldc + gn1] = acc11[r] * alpha + bv1;
    }
  } else {
    float bv0 = (bias && gn0 < N) ? bias[gn0] : 0.f;
    float bv1 = (bias && gn1 < N) ? bias[gn1] : 0.f;
    #pragma unroll
    for (int r = 0; r < 8; ++r) {
      if (gm0 + r < M) {
        if (gn0 < N) C[(size_t)(gm0 + r) * ldc + gn0] = acc00[r] * alpha + bv0;
        if (gn1 < N) C[(size_t)(gm0 + r) * ldc + gn1] = acc01[r] * alpha + bv1;
      }
      if (gm1 + r < M) {
        if (gn0 < N) C[(size_t)(gm1 + r) * ldc + gn0] = acc10[r] * alpha + bv0;
        if (gn1 < N) C[(size_t)(gm1 + r) * ldc + gn1] = acc11[r] * alpha + bv1;
      }
    }
  }
}

// ---------------------------------------------------------------------------
// Depthwise 3x3 stride-2 pool + LayerNorm; one wave per output token, 2 ch/lane.
// Output buffers have NTOKP(=832) token rows; rows >= 785 are zero-filled so
// downstream GEMMs can use K=832 without tail handling.
// wid = ((which*B + b)*NH + h)*NTOKP + l
// ---------------------------------------------------------------------------
__global__ __launch_bounds__(256)
void pool_ln_kernel(const float* __restrict__ qkv,
                    const float* __restrict__ wq, const float* __restrict__ wk,
                    const float* __restrict__ wv,
                    const float* __restrict__ gq, const float* __restrict__ bq,
                    const float* __restrict__ gk, const float* __restrict__ bk,
                    const float* __restrict__ gv, const float* __restrict__ bv,
                    float* __restrict__ qo, float* __restrict__ ko,
                    float* __restrict__ vo, int B, int nwaves)
{
  int wid  = (int)((blockIdx.x * blockDim.x + threadIdx.x) >> 5);
  int lane = threadIdx.x & 31;
  if (wid >= nwaves) return;

  int l  = wid % NTOKP; int t1 = wid / NTOKP;
  int h  = t1 % NHD;    int t2 = t1 / NHD;
  int b  = t2 % B;      int which = t2 / B;

  float* out = (which == 0) ? qo : (which == 1) ? ko : vo;
  const int c0 = lane * 2;
  size_t ob = ((size_t)((b * NHD + h) * NTOKP) + l) * HDD;

  if (l >= NTOK) {                       // zero padding rows (uniform per wave)
    out[ob + c0] = 0.f; out[ob + c0 + 1] = 0.f;
    return;
  }

  const float* w  = (which == 0) ? wq : (which == 1) ? wk : wv;
  const float* g  = (which == 0) ? gq : (which == 1) ? gk : gv;
  const float* be = (which == 0) ? bq : (which == 1) ? bk : bv;
  const size_t colbase = (size_t)which * DIMM + (size_t)h * HDD;

  float s0, s1;
  if (l == 0) {
    size_t base = ((size_t)b * NTOK_IN) * (3 * DIMM) + colbase;
    s0 = qkv[base + c0]; s1 = qkv[base + c0 + 1];
  } else {
    int p = l - 1, oi = p / GP, oj = p % GP;
    s0 = 0.f; s1 = 0.f;
    #pragma unroll
    for (int di = 0; di < 3; ++di)
      #pragma unroll
      for (int dj = 0; dj < 3; ++dj) {
        int ii = oi * 2 - 1 + di, jj = oj * 2 - 1 + dj;
        if ((unsigned)ii < 56u && (unsigned)jj < 56u) {
          size_t base = ((size_t)b * NTOK_IN + 1 + ii * 56 + jj) * (3 * DIMM) + colbase;
          float w0 = w[(di * 3 + dj) * HDD + c0];
          float w1 = w[(di * 3 + dj) * HDD + c0 + 1];
          s0 += qkv[base + c0] * w0;
          s1 += qkv[base + c0 + 1] * w1;
        }
      }
  }
  // LayerNorm over 64 channels (wave32 shuffle reduction, 2 ch per lane)
  float sum = s0 + s1;
  for (int off = 16; off; off >>= 1) sum += __shfl_xor(sum, off, 32);
  float mean = sum * (1.f / 64.f);
  float d0 = s0 - mean, d1 = s1 - mean;
  float vs = d0 * d0 + d1 * d1;
  for (int off = 16; off; off >>= 1) vs += __shfl_xor(vs, off, 32);
  float inv = rsqrtf(vs * (1.f / 64.f) + 1e-5f);

  out[ob + c0]     = d0 * inv * g[c0]     + be[c0];
  out[ob + c0 + 1] = d1 * inv * g[c0 + 1] + be[c0 + 1];
}

// ---------------------------------------------------------------------------
// Decomposed rel-pos bias tables (q pooled rows have NTOKP stride):
//   rel_h[bh,pix,k] = sum_c q[bh,1+pix,c] * rel_pos_h[ih-k+27, c]
//   rel_w[bh,pix,k] = sum_c q[bh,1+pix,c] * rel_pos_w[iw-k+27, c]
// ---------------------------------------------------------------------------
__global__ __launch_bounds__(256)
void rel_bias_kernel(const float* __restrict__ qp, const float* __restrict__ rph,
                     const float* __restrict__ rpw, float* __restrict__ relh,
                     float* __restrict__ relw, int total)
{
  int i = blockIdx.x * blockDim.x + threadIdx.x;
  if (i >= total) return;
  int k   = i % GP;
  int pix = (i / GP) % NPIX;
  int bh  = i / (GP * NPIX);
  int ih = pix / GP, iw = pix % GP;
  const float* qv = qp + ((size_t)bh * NTOKP + 1 + pix) * HDD;
  const float* rh = rph + (size_t)(ih - k + GP - 1) * HDD;
  const float* rw = rpw + (size_t)(iw - k + GP - 1) * HDD;
  float sh_ = 0.f, sw_ = 0.f;
  #pragma unroll 8
  for (int c = 0; c < HDD; ++c) {
    float q = qv[c];
    sh_ += q * rh[c];
    sw_ += q * rw[c];
  }
  relh[i] = sh_;
  relw[i] = sw_;
}

// ---------------------------------------------------------------------------
// Softmax with fused rel-pos bias add; one wave per row; row kept in registers.
// attn has leading dim NTOKP(=832); columns 785..831 are written as zeros so
// the following attn@v GEMM can run K=832 unguarded.
// ---------------------------------------------------------------------------
#define SM_ITERS (NTOKP / 32)   // 26

__global__ __launch_bounds__(256)
void softmax_bias_kernel(float* __restrict__ attn, const float* __restrict__ relh,
                         const float* __restrict__ relw, int rows)
{
  __shared__ float sh[8][64];
  int wid  = (int)((blockIdx.x * blockDim.x + threadIdx.x) >> 5);
  int lane = threadIdx.x & 31;
  if (wid >= rows) return;
  int bh = wid / NTOK, row = wid % NTOK;
  int wv = (threadIdx.x >> 5) & 7;
  float* bh_lds = sh[wv];

  if (row > 0 && lane < GP) {
    int qp = row - 1;
    bh_lds[lane]      = relh[((size_t)bh * NPIX + qp) * GP + lane];
    bh_lds[32 + lane] = relw[((size_t)bh * NPIX + qp) * GP + lane];
  }

  float* rowptr = attn + (size_t)wid * NTOKP;
  float v[SM_ITERS];
  float maxv = -3.4e38f;
  #pragma unroll
  for (int it = 0; it < SM_ITERS; ++it) {
    int j = lane + it * 32;
    float val = -3.4e38f;
    if (j < NTOK) {
      val = rowptr[j];
      if (row > 0 && j > 0) {
        int p = j - 1;
        val += bh_lds[p / GP] + bh_lds[32 + p % GP];
      }
    }
    v[it] = val;
    maxv = fmaxf(maxv, val);
  }
  for (int off = 16; off; off >>= 1) maxv = fmaxf(maxv, __shfl_xor(maxv, off, 32));
  float sum = 0.f;
  #pragma unroll
  for (int it = 0; it < SM_ITERS; ++it) {
    int j = lane + it * 32;
    if (j < NTOK) { float e = __expf(v[it] - maxv); v[it] = e; sum += e; }
  }
  for (int off = 16; off; off >>= 1) sum += __shfl_xor(sum, off, 32);
  float inv = 1.f / sum;
  #pragma unroll
  for (int it = 0; it < SM_ITERS; ++it) {
    int j = lane + it * 32;
    rowptr[j] = (j < NTOK) ? v[it] * inv : 0.f;   // zero K-padding cols 785..831
  }
}

// ---------------------------------------------------------------------------
// out = attn@v + q residual, relayout [b,h,l,d] -> [b,l,h*64+d]
// O is compact [bh][785][64]; qp has NTOKP row stride.
// ---------------------------------------------------------------------------
__global__ __launch_bounds__(256)
void merge_residual_kernel(const float* __restrict__ O, const float* __restrict__ qp,
                           float* __restrict__ X2, int total)
{
  int i = blockIdx.x * blockDim.x + threadIdx.x;
  if (i >= total) return;
  int d  = i & (HDD - 1);
  int l  = (i >> 6) % NTOK;
  int bh = i / (HDD * NTOK);
  int h  = bh % NHD;
  int b  = bh / NHD;
  float val = O[i] + qp[((size_t)bh * NTOKP + l) * HDD + d];
  X2[(size_t)(b * NTOK + l) * DIMM + h * HDD + d] = val;
}

// ---------------------------------------------------------------------------
extern "C" void kernel_launch(void* const* d_in, const int* in_sizes, int n_in,
                              void* d_out, int out_size, void* d_ws, size_t ws_size,
                              hipStream_t stream)
{
  const float* x      = (const float*)d_in[0];
  const float* qkv_w  = (const float*)d_in[1];
  const float* pqw    = (const float*)d_in[2];
  const float* pkw    = (const float*)d_in[3];
  const float* pvw    = (const float*)d_in[4];
  const float* nq_g   = (const float*)d_in[5];
  const float* nq_b   = (const float*)d_in[6];
  const float* nk_g   = (const float*)d_in[7];
  const float* nk_b   = (const float*)d_in[8];
  const float* nv_g   = (const float*)d_in[9];
  const float* nv_b   = (const float*)d_in[10];
  const float* rph    = (const float*)d_in[11];
  const float* rpw    = (const float*)d_in[12];
  const float* proj_w = (const float*)d_in[13];
  const float* proj_b = (const float*)d_in[14];
  (void)n_in; (void)out_size; (void)ws_size;

  const int B  = in_sizes[0] / (NTOK_IN * DIMM);   // 8
  const int M0 = B * NTOK_IN;                      // 25096
  const int BH = B * NHD;                          // 96

  float* ws = (float*)d_ws;
  size_t off = 0;
  float* qkv_o = ws + off; off += (size_t)M0 * 3 * DIMM;
  float* qp    = ws + off; off += (size_t)BH * NTOKP * HDD;
  float* kp    = ws + off; off += (size_t)BH * NTOKP * HDD;
  float* vp    = ws + off; off += (size_t)BH * NTOKP * HDD;
  float* relh  = ws + off; off += (size_t)BH * NPIX * GP;
  float* relw  = ws + off; off += (size_t)BH * NPIX * GP;
  float* attn  = ws + off; off += (size_t)BH * NTOK * NTOKP;   // ld = NTOKP
  float* Ob    = ws + off; off += (size_t)BH * NTOK * HDD;
  float* X2    = ws + off; off += (size_t)B * NTOK * DIMM;

  // 1) QKV GEMM: [25096,768] x [768,2304]   (K=768)
  {
    dim3 g((3 * DIMM + BNT - 1) / BNT, (M0 + BMT - 1) / BMT, 1);
    wmma_gemm_kernel<true><<<g, 256, 0, stream>>>(x, qkv_w, qkv_o, nullptr,
        M0, 3 * DIMM, DIMM, DIMM, 3 * DIMM, 3 * DIMM, 0LL, 0LL, 0LL, 1.f);
  }
  // 2) depthwise pool + LayerNorm for q,k,v (with zero-padded rows 785..831)
  {
    int nwaves = 3 * BH * NTOKP;
    int blocks = (nwaves * 32 + 255) / 256;
    pool_ln_kernel<<<blocks, 256, 0, stream>>>(qkv_o, pqw, pkw, pvw,
        nq_g, nq_b, nk_g, nk_b, nv_g, nv_b, qp, kp, vp, B, nwaves);
  }
  // 3) rel-pos bias tables
  {
    int total = BH * NPIX * GP;
    rel_bias_kernel<<<(total + 255) / 256, 256, 0, stream>>>(qp, rph, rpw, relh, relw, total);
  }
  // 4) scores = scale * q @ k^T   (B matrix = k stored [N,K], K=64)
  {
    dim3 g((NTOK + BNT - 1) / BNT, (NTOK + BMT - 1) / BMT, BH);
    wmma_gemm_kernel<false><<<g, 256, 0, stream>>>(qp, kp, attn, nullptr,
        NTOK, NTOK, HDD, HDD, HDD, NTOKP,
        (long long)NTOKP * HDD, (long long)NTOKP * HDD, (long long)NTOK * NTOKP, 0.125f);
  }
  // 5) softmax with fused rel-pos bias (also zeroes K-padding columns)
  {
    int rows = BH * NTOK;
    int blocks = (rows * 32 + 255) / 256;
    softmax_bias_kernel<<<blocks, 256, 0, stream>>>(attn, relh, relw, rows);
  }
  // 6) O = attn @ v   (K padded to 832; pad cols of attn and pad rows of v are 0)
  {
    dim3 g((HDD + BNT - 1) / BNT, (NTOK + BMT - 1) / BMT, BH);
    wmma_gemm_kernel<true><<<g, 256, 0, stream>>>(attn, vp, Ob, nullptr,
        NTOK, HDD, NTOKP, NTOKP, HDD, HDD,
        (long long)NTOK * NTOKP, (long long)NTOKP * HDD, (long long)NTOK * HDD, 1.f);
  }
  // 7) residual + relayout
  {
    int total = BH * NTOK * HDD;
    merge_residual_kernel<<<(total + 255) / 256, 256, 0, stream>>>(Ob, qp, X2, total);
  }
  // 8) output projection with bias -> d_out  (K=768)
  {
    dim3 g((DIMM + BNT - 1) / BNT, (B * NTOK + BMT - 1) / BMT, 1);
    wmma_gemm_kernel<true><<<g, 256, 0, stream>>>(X2, proj_w, (float*)d_out, proj_b,
        B * NTOK, DIMM, DIMM, DIMM, DIMM, DIMM, 0LL, 0LL, 0LL, 1.f);
  }
}
